// VisualConvolution_81570018886287
// MI455X (gfx1250) — matrix operand order repaired and checked by reading, not compiled, not run on hardware
//
#include <hip/hip_runtime.h>
#include <hip/hip_bf16.h>

// ---------------------------------------------------------------------------
// Problem constants (from the reference)
// ---------------------------------------------------------------------------
#define BATCH   512
#define KNBR    36
#define DDIM    1024
#define NKER    4
#define OUTDIM  1024
#define MROWS   (BATCH * KNBR)        // 18432

typedef __attribute__((ext_vector_type(16))) __bf16 v16bf;
typedef __attribute__((ext_vector_type(8)))  float  v8f;
typedef __attribute__((ext_vector_type(4)))  float  v4f;

// Pack the high 16 bits (truncated bf16) of two f32 into one dword: 1 v_perm_b32.
static __device__ __forceinline__ unsigned pack_hi16(unsigned u1, unsigned u0) {
    return __builtin_amdgcn_perm(u1, u0, 0x07060302u);   // {u1.hi16, u0.hi16}
}

// ---------------------------------------------------------------------------
// Kernel 1: ws[m, n] = sum_j softmax_n( exp(-0.5 (rrw[m,j]-mean_n)^2 / (eps+prec_n^2)) )
// ---------------------------------------------------------------------------
__global__ void ws_kernel(const float* __restrict__ rrw,
                          const float* __restrict__ mean,
                          const float* __restrict__ prec,
                          float* __restrict__ ws) {
    int m = blockIdx.x * blockDim.x + threadIdx.x;   // 0 .. MROWS-1
    if (m >= MROWS) return;
    float mu0 = mean[0], mu1 = mean[1], mu2 = mean[2], mu3 = mean[3];
    float p0 = prec[0], p1 = prec[1], p2 = prec[2], p3 = prec[3];
    float i0 = -0.5f / (1e-14f + p0 * p0);
    float i1 = -0.5f / (1e-14f + p1 * p1);
    float i2 = -0.5f / (1e-14f + p2 * p2);
    float i3 = -0.5f / (1e-14f + p3 * p3);
    const float* row = rrw + (size_t)m * KNBR;
    float s0 = 0.f, s1 = 0.f, s2 = 0.f, s3 = 0.f;
    #pragma unroll 4
    for (int j = 0; j < KNBR; ++j) {
        float x = row[j];
        float d0 = x - mu0, d1 = x - mu1, d2 = x - mu2, d3 = x - mu3;
        float e0 = __expf(d0 * d0 * i0);
        float e1 = __expf(d1 * d1 * i1);
        float e2 = __expf(d2 * d2 * i2);
        float e3 = __expf(d3 * d3 * i3);
        float r = 1.0f / (e0 + e1 + e2 + e3);
        s0 += e0 * r; s1 += e1 * r; s2 += e2 * r; s3 += e3 * r;
    }
    float* o = ws + (size_t)m * NKER;
    o[0] = s0; o[1] = s1; o[2] = s2; o[3] = s3;
}

// ---------------------------------------------------------------------------
// Kernel 2: split W (f32 [1024,1024]) into bf16 hi/lo (truncate + v_perm).
// One dword (= 2 bf16) per thread per array.
// ---------------------------------------------------------------------------
__global__ void wsplit_kernel(const float* __restrict__ W,
                              unsigned* __restrict__ wh,
                              unsigned* __restrict__ wl) {
    int p = blockIdx.x * blockDim.x + threadIdx.x;       // pair index
    if (p >= OUTDIM * DDIM / 2) return;
    float x0 = W[2 * p], x1 = W[2 * p + 1];
    unsigned u0 = __builtin_bit_cast(unsigned, x0);
    unsigned u1 = __builtin_bit_cast(unsigned, x1);
    wh[p] = pack_hi16(u1, u0);
    float r0 = x0 - __builtin_bit_cast(float, u0 & 0xffff0000u);
    float r1 = x1 - __builtin_bit_cast(float, u1 & 0xffff0000u);
    wl[p] = pack_hi16(__builtin_bit_cast(unsigned, r1),
                      __builtin_bit_cast(unsigned, r0));
}

// ---------------------------------------------------------------------------
// Kernel 3: GEMM  out[m,c] = ws[m, c>>8] * sum_d X[m,d] * Wmat[c,d]
// 3xBF16-split matmul on v_wmma_f32_16x16x32_bf16, f32 accumulation.
//
// Per-wave tile: 64 rows (4 row-groups) x 128 cols (8 col-tiles)
//   -> 32 accumulator tiles (256 VGPRs), 96 WMMAs per K-step,
//      A 8KB + B 16KB per K-step  =>  ~250 B/WMMA (balanced vs vL1).
// Block = 4 waves stacked on M (256 rows); all waves share B frags via WGP$.
// Grid: x = MROWS/256 = 72, y = OUTDIM/128 = 8.
// ---------------------------------------------------------------------------
__global__ __launch_bounds__(128) void gemm_kernel(
    const float*  __restrict__ X,    // [MROWS, DDIM] f32
    const __bf16* __restrict__ Wh,   // [OUTDIM, DDIM] bf16 hi
    const __bf16* __restrict__ Wl,   // [OUTDIM, DDIM] bf16 lo
    const float*  __restrict__ ws,   // [MROWS, NKER]
    float*        __restrict__ out)  // [MROWS, OUTDIM]
{
    const int lane = threadIdx.x & 31;
    const int wave = threadIdx.x >> 5;             // 0..3
    const int l15  = lane & 15;
    const int half = lane >> 4;                    // 0 | 1

    const int m0 = blockIdx.x * 256 + wave * 64;   // this wave's 64 rows
    const int c0 = blockIdx.y * 128;               // this wave's 128 cols

    // A-fragment bases per row-group g: row = m0 + g*16 + l15;
    //   elems 0-15 at d = kk + half*16 + e
    const float* arow[4];
    #pragma unroll
    for (int g = 0; g < 4; ++g)
        arow[g] = X + (size_t)(m0 + g * 16 + l15) * DDIM + half * 16;

    v8f acc[8][4];                                 // [col-tile][row-group]
    #pragma unroll
    for (int t = 0; t < 8; ++t)
        #pragma unroll
        for (int g = 0; g < 4; ++g)
            acc[t][g] = v8f{};

    union V16 { v16bf v; unsigned u[8]; };

    for (int kk = 0; kk < DDIM; kk += 32) {
        // ---- load + truncate-split 4 A fragments ----
        V16 ahi[4], alo[4];
        #pragma unroll
        for (int g = 0; g < 4; ++g) {
            v4f av[4];
            av[0] = *(const v4f*)(arow[g] + kk);
            av[1] = *(const v4f*)(arow[g] + kk + 4);
            av[2] = *(const v4f*)(arow[g] + kk + 8);
            av[3] = *(const v4f*)(arow[g] + kk + 12);
            #pragma unroll
            for (int p = 0; p < 8; ++p) {          // bf16 pairs
                float x0 = av[p >> 1][(p & 1) * 2];
                float x1 = av[p >> 1][(p & 1) * 2 + 1];
                unsigned u0 = __builtin_bit_cast(unsigned, x0);
                unsigned u1 = __builtin_bit_cast(unsigned, x1);
                ahi[g].u[p] = pack_hi16(u1, u0);
                float r0 = x0 - __builtin_bit_cast(float, u0 & 0xffff0000u);
                float r1 = x1 - __builtin_bit_cast(float, u1 & 0xffff0000u);
                alo[g].u[p] = pack_hi16(__builtin_bit_cast(unsigned, r1),
                                        __builtin_bit_cast(unsigned, r0));
            }
        }

        // ---- 8 col-tiles: one hi + one lo B fragment each, 12 WMMAs ----
        #pragma unroll
        for (int t = 0; t < 8; ++t) {
            size_t woff = (size_t)(c0 + t * 16 + l15) * DDIM + kk + half * 16;
            v16bf bhi = *(const v16bf*)(Wh + woff);   // 32B aligned
            v16bf blo = *(const v16bf*)(Wl + woff);
            #pragma unroll
            for (int g = 0; g < 4; ++g) {
                // (ahi+alo)(bhi+blo) ~= ahi*bhi + alo*bhi + ahi*blo
                acc[t][g] = __builtin_amdgcn_wmma_f32_16x16x32_bf16(
                    false, ahi[g].v, false, bhi, (short)0, acc[t][g], false, false);
                acc[t][g] = __builtin_amdgcn_wmma_f32_16x16x32_bf16(
                    false, alo[g].v, false, bhi, (short)0, acc[t][g], false, false);
                acc[t][g] = __builtin_amdgcn_wmma_f32_16x16x32_bf16(
                    false, ahi[g].v, false, blo, (short)0, acc[t][g], false, false);
            }
        }
    }

    // ---- scale by ws and store.  acc[t][g][r] -> (M = m0+g*16+r+half*8, N = c0+t*16+l15)
    const int n = c0 >> 8;                 // kernel-block index (wave-constant)
    float sc[4][8];
    #pragma unroll
    for (int g = 0; g < 4; ++g)
        #pragma unroll
        for (int r = 0; r < 8; ++r)
            sc[g][r] = ws[(size_t)(m0 + g * 16 + r + half * 8) * NKER + n];

    #pragma unroll
    for (int t = 0; t < 8; ++t) {
        int c = c0 + t * 16 + l15;
        #pragma unroll
        for (int g = 0; g < 4; ++g)
            #pragma unroll
            for (int r = 0; r < 8; ++r) {
                int m = m0 + g * 16 + r + half * 8;
                out[(size_t)m * OUTDIM + c] = acc[t][g][r] * sc[g][r];
            }
    }
}

// ---------------------------------------------------------------------------
// Launch.  Scratch layout in d_ws:
//   [0)                ws   : MROWS*4 floats        (294,912 B)
//   [294912)           w_hi : 1024*1024 bf16        (2 MiB)
//   [294912+2MiB)      w_lo : 1024*1024 bf16        (2 MiB)
// Total ~4.5 MB.
// ---------------------------------------------------------------------------
extern "C" void kernel_launch(void* const* d_in, const int* in_sizes, int n_in,
                              void* d_out, int out_size, void* d_ws, size_t ws_size,
                              hipStream_t stream) {
    const float* feat = (const float*)d_in[0];   // [B,K,D]
    const float* rrw  = (const float*)d_in[1];   // [B,K,K]
    const float* mean = (const float*)d_in[2];   // [4]
    const float* prec = (const float*)d_in[3];   // [4]
    const float* W    = (const float*)d_in[4];   // [4,256,1024] == [1024,1024]
    float* out = (float*)d_out;                  // [B,K,1024]

    float*  ws = (float*)d_ws;
    __bf16* wh = (__bf16*)((char*)d_ws + (size_t)MROWS * NKER * sizeof(float));
    __bf16* wl = wh + (size_t)OUTDIM * DDIM;

    ws_kernel<<<(MROWS + 255) / 256, 256, 0, stream>>>(rrw, mean, prec, ws);
    wsplit_kernel<<<(OUTDIM * DDIM / 2 + 255) / 256, 256, 0, stream>>>(
        W, (unsigned*)wh, (unsigned*)wl);

    dim3 grid(MROWS / 256, OUTDIM / 128);        // (72, 8)
    gemm_kernel<<<grid, 128, 0, stream>>>(feat, wh, wl, ws, out);
}
